// TopoGatedMLPGIN_48120813584812
// MI455X (gfx1250) — compile-verified, matrix operand order checked by reading, not run on hardware
//
#include <hip/hip_runtime.h>
#include <hip/hip_bf16.h>

typedef __attribute__((ext_vector_type(16))) __bf16 v16bf;
typedef __attribute__((ext_vector_type(8)))  float  v8f;
typedef __attribute__((ext_vector_type(4)))  unsigned int u32x4;
typedef __attribute__((ext_vector_type(8)))  int i32x8;
typedef __attribute__((ext_vector_type(4)))  int i32x4;

#define TOPO_DIM 32
#define IN_DIM   64
#define HID_DIM  128
#define NCLS     10
#define FRAG_ELEMS 512   // one B fragment: 32 lanes x 16 bf16

#if __has_builtin(__builtin_amdgcn_tensor_load_to_lds)
#define HAVE_TDM 1
#else
#define HAVE_TDM 0
#endif

// K index pattern for 16-bit B fragments (ISA 7.12.2) - used only by the
// one-time weight swizzle (B is a column gather from row-major W).
__device__ __forceinline__ int kpat(int v, int hi) {
    return (v < 4) ? (8 * hi + 2 * v) : (16 + 8 * hi + 2 * (v - 4));
}

__device__ __forceinline__ v8f wmma_bf16(v16bf a, v16bf b, v8f c) {
    return __builtin_amdgcn_wmma_f32_16x16x32_bf16(false, a, false, b, (short)0, c, false, false);
}

// CDNA5 hardware tanh (TRANS op). v_nop covers the 1-op TRANS result hazard
// since the compiler cannot see through the asm block.
__device__ __forceinline__ float fast_tanh(float x) {
    float r;
    asm("v_tanh_f32 %0, %1\n\tv_nop" : "=v"(r) : "v"(x));
    return r;
}
__device__ __forceinline__ float fast_sigmoid(float x) {
    return fmaf(0.5f, fast_tanh(0.5f * x), 0.5f);
}

__device__ __forceinline__ void wait_tensor0() {
#if __has_builtin(__builtin_amdgcn_s_wait_tensorcnt)
    __builtin_amdgcn_s_wait_tensorcnt(0);
#else
    asm volatile("s_wait_tensorcnt 0x0" ::: "memory");
#endif
}

#if HAVE_TDM
// 1-D TDM tile load: ne bf16 elements global -> LDS (ne <= 65535).
// D# layout per CDNA5 ISA.
__device__ __forceinline__ void tdm_load_bf16_1d(unsigned int lds_off, const void* gsrc,
                                                 unsigned int ne) {
    unsigned long long ga = (unsigned long long)(uintptr_t)gsrc;
    u32x4 g0;
    g0[0] = 1u;                                                    // count=1 (valid user D#)
    g0[1] = lds_off;                                               // lds_addr
    g0[2] = (unsigned int)ga;                                      // global_addr[31:0]
    g0[3] = (unsigned int)((ga >> 32) & 0x1FFFFFFu) | (2u << 30);  // addr[56:32] | type=2
    i32x8 g1;
    g1[0] = (int)(1u << 16);                           // data_size=1 -> 2 bytes/elem
    g1[1] = (int)((ne & 0xFFFFu) << 16);               // tensor_dim0[15:0]
    g1[2] = (int)(((ne >> 16) & 0xFFFFu) | (1u << 16));// tensor_dim0[31:16] | tensor_dim1=1
    g1[3] = (int)((ne & 0xFFFFu) << 16);               // tile_dim0 = ne
    g1[4] = 1;                                         // tile_dim1=1, tile_dim2=0
    g1[5] = (int)ne;                                   // tensor_dim0_stride[31:0]
    g1[6] = 0;
    g1[7] = 0;
    i32x4 z = {0, 0, 0, 0};
#if defined(__clang_major__) && (__clang_major__ >= 23)
    i32x8 z8 = {0, 0, 0, 0, 0, 0, 0, 0};
    __builtin_amdgcn_tensor_load_to_lds(g0, g1, z, z, z8, 0);
#else
    __builtin_amdgcn_tensor_load_to_lds(g0, g1, z, z, 0);
#endif
}
#endif

// stage `ne` bf16 elements of pre-swizzled weight fragments into LDS
__device__ __forceinline__ void stage_weights(__bf16* sDst, const unsigned short* gSrc,
                                              unsigned int ne, int tid, int nthreads) {
#if HAVE_TDM
    if (tid == 0) {
        tdm_load_bf16_1d((unsigned int)(uintptr_t)sDst, gSrc, ne);
        wait_tensor0();
    }
#else
    const uint4* s = (const uint4*)gSrc;
    uint4* d = (uint4*)sDst;
    for (unsigned int i = tid; i < ne / 8; i += nthreads) d[i] = s[i];
#endif
}

// one B fragment = one aligned 32B LDS vector load
__device__ __forceinline__ v16bf load_frag(const __bf16* F, int f, int lane) {
    return ((const v16bf*)F)[f * 32 + lane];
}

__device__ __forceinline__ float4 san4(float4 v) {
    v.x = __builtin_isfinite(v.x) ? v.x : 0.0f;
    v.y = __builtin_isfinite(v.y) ? v.y : 0.0f;
    v.z = __builtin_isfinite(v.z) ? v.z : 0.0f;
    v.w = __builtin_isfinite(v.w) ? v.w : 0.0f;
    return v;
}

// 16 f32 -> v16bf (compiler emits packed v_cvt_pk_bf16_f32)
__device__ __forceinline__ v16bf pack16(const float* f) {
    v16bf r;
#pragma unroll
    for (int i = 0; i < 16; ++i) r[i] = (__bf16)f[i];
    return r;
}

// A fragment (16 rows x 32 K) from row-major f32 matrix, optional nan_to_num.
// Per-lane A data is two contiguous 8-element runs -> 4x float4 loads.
__device__ __forceinline__ v16bf load_a_global(const float* __restrict__ base, int ld,
                                               int row0, int k0, int lane, int nrows, bool san) {
    int m = lane & 15, hi = lane >> 4;
    int r = row0 + m; if (r >= nrows) r = nrows - 1;
    const float* rp = base + (size_t)r * ld + k0 + 8 * hi;
    float4 q[4];
    q[0] = *(const float4*)(rp + 0);
    q[1] = *(const float4*)(rp + 4);
    q[2] = *(const float4*)(rp + 16);
    q[3] = *(const float4*)(rp + 20);
    if (san) {
        q[0] = san4(q[0]); q[1] = san4(q[1]); q[2] = san4(q[2]); q[3] = san4(q[3]);
    }
    return pack16((const float*)q);
}

// A fragment = elementwise sum of two row-major f32 matrices (GIN: x + agg).
__device__ __forceinline__ v16bf load_a_sum(const float* __restrict__ A0, const float* __restrict__ A1,
                                            int ld, int row0, int k0, int lane, int nrows) {
    int m = lane & 15, hi = lane >> 4;
    int r = row0 + m; if (r >= nrows) r = nrows - 1;
    const float* p0 = A0 + (size_t)r * ld + k0 + 8 * hi;
    const float* p1 = A1 + (size_t)r * ld + k0 + 8 * hi;
    float4 q[4];
#pragma unroll
    for (int j = 0; j < 4; ++j) {
        int off = (j >> 1) * 16 + (j & 1) * 4;
        float4 u = *(const float4*)(p0 + off);
        float4 w = *(const float4*)(p1 + off);
        q[j] = make_float4(u.x + w.x, u.y + w.y, u.z + w.z, u.w + w.w);
    }
    return pack16((const float*)q);
}

// A fragment from per-wave LDS staging tile (row-major 16 x ld bf16):
// two ds_load_b128 per fragment.
__device__ __forceinline__ v16bf load_a_lds(const __bf16* S, int ld, int k0, int lane) {
    int m = lane & 15, hi = lane >> 4;
    const __bf16* rp = S + m * ld + k0 + 8 * hi;
    v16bf a;
    ((uint4*)&a)[0] = *(const uint4*)(rp + 0);
    ((uint4*)&a)[1] = *(const uint4*)(rp + 16);
    return a;
}

// ---------------------------------------------------------------------------
// Kernel 0: pre-swizzle a row-major f32 weight [K][Nout] into bf16 WMMA
// B-fragment order: frag f = kt*(Nout/16)+nt, 32 lanes x 16 contiguous bf16.
// ---------------------------------------------------------------------------
__global__ void __launch_bounds__(32)
swizzle_w_kernel(const float* __restrict__ W, int Nout, unsigned short* __restrict__ dst) {
    int f = blockIdx.x;
    int NT = Nout >> 4;
    int kt = f / NT, nt = f % NT;
    int lane = threadIdx.x;
    int n = nt * 16 + (lane & 15), hi = lane >> 4;
    __attribute__((aligned(32))) __bf16 tmp[16];
#pragma unroll
    for (int v = 0; v < 8; ++v) {
        int kk = kt * 32 + kpat(v, hi);
        tmp[2 * v]     = (__bf16)W[(size_t)kk * Nout + n];
        tmp[2 * v + 1] = (__bf16)W[(size_t)(kk + 1) * Nout + n];
    }
    *(v16bf*)((__bf16*)dst + ((size_t)f * 32 + lane) * 16) = *(v16bf*)tmp;
}

// ---------------------------------------------------------------------------
// Kernel 1: gated MLP on topo features + concat with x -> xc [N,128]
// wfr: 32 pre-swizzled frags = [w_lin(8) | w_gate(8) | w_tout(16)]
// ---------------------------------------------------------------------------
__global__ void __launch_bounds__(256)
gated_topo_kernel(const float* __restrict__ x, const float* __restrict__ topo,
                  const unsigned short* __restrict__ wfr,
                  const float* __restrict__ b_lin, const float* __restrict__ b_gate,
                  const float* __restrict__ b_tout,
                  float* __restrict__ xc, int Nn) {
    extern __shared__ char smem[];
    __bf16* sFr = (__bf16*)smem;                       // 32 frags = 16384 bf16
    __bf16* sStageAll = sFr + 32 * FRAG_ELEMS;         // 8 waves * 16*128
    float* sBlin  = (float*)(sStageAll + 8 * 16 * HID_DIM);
    float* sBgate = sBlin + HID_DIM;
    float* sBtout = sBgate + HID_DIM;

    int tid = threadIdx.x;
    stage_weights(sFr, wfr, 32 * FRAG_ELEMS, tid, blockDim.x);
    for (int i = tid; i < HID_DIM; i += blockDim.x) { sBlin[i] = b_lin[i]; sBgate[i] = b_gate[i]; }
    for (int i = tid; i < IN_DIM; i += blockDim.x) sBtout[i] = b_tout[i];
    __syncthreads();

    int lane = tid & 31, wave = tid >> 5;
    int m = lane & 15, hi = lane >> 4;
    __bf16* sStage = sStageAll + wave * 16 * HID_DIM;

    int ntiles = (Nn + 15) >> 4;
    int gw = blockIdx.x * (blockDim.x >> 5) + wave;
    int nw = gridDim.x * (blockDim.x >> 5);

    for (int t = gw; t < ntiles; t += nw) {
        int row0 = t << 4;
        v16bf atf = load_a_global(topo, TOPO_DIM, row0, 0, lane, Nn, true);
        // h = tanh(tf@w_lin+b), g = sigmoid(tf@w_gate+b); stage = bf16(h*g)
#pragma unroll
        for (int nt = 0; nt < 8; ++nt) {
            v16bf bl = load_frag(sFr, nt, lane);
            v16bf bg = load_frag(sFr, 8 + nt, lane);
            v8f ch = {}; ch = wmma_bf16(atf, bl, ch);
            v8f cg = {}; cg = wmma_bf16(atf, bg, cg);
#pragma unroll
            for (int v = 0; v < 8; ++v) {
                int n = nt * 16 + m;
                int mr = v + 8 * hi;
                float hv = fast_tanh(ch[v] + sBlin[n]);
                float gv = fast_sigmoid(cg[v] + sBgate[n]);
                sStage[mr * HID_DIM + n] = (__bf16)(hv * gv);
            }
        }
        // topo_out[16,64] = stage[16,128] @ w_tout[128,64] + b_tout
#pragma unroll
        for (int ot = 0; ot < 4; ++ot) {
            v8f acc = {};
#pragma unroll
            for (int kt = 0; kt < 4; ++kt) {
                v16bf a2 = load_a_lds(sStage, HID_DIM, kt * 32, lane);
                v16bf b2 = load_frag(sFr, 16 + kt * 4 + ot, lane);
                acc = wmma_bf16(a2, b2, acc);
            }
#pragma unroll
            for (int v = 0; v < 8; ++v) {
                int n = ot * 16 + m;
                int r = row0 + v + 8 * hi;
                if (r < Nn) xc[(size_t)r * HID_DIM + IN_DIM + n] = acc[v] + sBtout[n];
            }
        }
        // xc[:, :64] = x
        for (int i = lane; i < 256; i += 32) {
            int mr = i >> 4, c4 = i & 15;
            int r = row0 + mr;
            if (r < Nn) {
                float4 v4 = ((const float4*)(x + (size_t)r * IN_DIM))[c4];
                ((float4*)(xc + (size_t)r * HID_DIM))[c4] = v4;
            }
        }
    }
}

// ---------------------------------------------------------------------------
// Kernel 2: edge scatter  agg[dst] += feat[src]   (one wave = one 512B row)
// ---------------------------------------------------------------------------
__global__ void __launch_bounds__(256)
edge_scatter_kernel(const float* __restrict__ feat, const int* __restrict__ src,
                    const int* __restrict__ dst, float* __restrict__ agg, int E) {
    long long stride = (long long)gridDim.x * blockDim.x;
    long long total = (long long)E * 32;
    for (long long i = (long long)blockIdx.x * blockDim.x + threadIdx.x; i < total; i += stride) {
        int e = (int)(i >> 5);
        int c = (int)(i & 31);
        long long ni = i + stride;
        if (ni < total) {
            int ne = (int)(ni >> 5);
            __builtin_prefetch(feat + (size_t)src[ne] * HID_DIM, 0, 1);  // global_prefetch_b8
        }
        int s = src[e], d = dst[e];
        float4 v = ((const float4*)(feat + (size_t)s * HID_DIM))[c];
        float* drow = agg + (size_t)d * HID_DIM + (size_t)c * 4;
        atomicAdd(drow + 0, v.x);
        atomicAdd(drow + 1, v.y);
        atomicAdd(drow + 2, v.z);
        atomicAdd(drow + 3, v.w);
    }
}

// ---------------------------------------------------------------------------
// Kernel 3: GIN MLP  out = relu(relu((xin+agg) @ w1 + b1) @ w2 + b2)
// wfr: 64 pre-swizzled frags = [w1(32) | w2(32)]
// ---------------------------------------------------------------------------
__global__ void __launch_bounds__(256)
gin_mlp_kernel(const float* __restrict__ xin, const float* __restrict__ agg,
               const unsigned short* __restrict__ wfr,
               const float* __restrict__ b1, const float* __restrict__ b2,
               float* __restrict__ out, int Nn) {
    extern __shared__ char smem[];
    __bf16* sFr = (__bf16*)smem;                       // 64 frags = 32768 bf16 (64KB)
    __bf16* sStageAll = sFr + 64 * FRAG_ELEMS;         // 8 * 16*128
    float* sB1 = (float*)(sStageAll + 8 * 16 * HID_DIM);
    float* sB2 = sB1 + HID_DIM;

    int tid = threadIdx.x;
    stage_weights(sFr, wfr, 64 * FRAG_ELEMS, tid, blockDim.x);
    for (int i = tid; i < HID_DIM; i += blockDim.x) { sB1[i] = b1[i]; sB2[i] = b2[i]; }
    __syncthreads();

    int lane = tid & 31, wave = tid >> 5;
    int m = lane & 15, hi = lane >> 4;
    __bf16* sStage = sStageAll + wave * 16 * HID_DIM;

    int ntiles = (Nn + 15) >> 4;
    int gw = blockIdx.x * (blockDim.x >> 5) + wave;
    int nw = gridDim.x * (blockDim.x >> 5);

    for (int t = gw; t < ntiles; t += nw) {
        int row0 = t << 4;
        v16bf a[4];
#pragma unroll
        for (int kt = 0; kt < 4; ++kt)
            a[kt] = load_a_sum(xin, agg, HID_DIM, row0, kt * 32, lane, Nn);
        // layer 1 -> LDS stage (bf16)
#pragma unroll
        for (int nt = 0; nt < 8; ++nt) {
            v8f acc = {};
#pragma unroll
            for (int kt = 0; kt < 4; ++kt)
                acc = wmma_bf16(a[kt], load_frag(sFr, kt * 8 + nt, lane), acc);
#pragma unroll
            for (int v = 0; v < 8; ++v) {
                int n = nt * 16 + m, mr = v + 8 * hi;
                sStage[mr * HID_DIM + n] = (__bf16)fmaxf(acc[v] + sB1[n], 0.0f);
            }
        }
        // layer 2
#pragma unroll
        for (int kt = 0; kt < 4; ++kt)
            a[kt] = load_a_lds(sStage, HID_DIM, kt * 32, lane);
#pragma unroll
        for (int nt = 0; nt < 8; ++nt) {
            v8f acc = {};
#pragma unroll
            for (int kt = 0; kt < 4; ++kt)
                acc = wmma_bf16(a[kt], load_frag(sFr, 32 + kt * 8 + nt, lane), acc);
#pragma unroll
            for (int v = 0; v < 8; ++v) {
                int n = nt * 16 + m;
                int r = row0 + v + 8 * hi;
                if (r < Nn) out[(size_t)r * HID_DIM + n] = fmaxf(acc[v] + sB2[n], 0.0f);
            }
        }
    }
}

// ---------------------------------------------------------------------------
// Kernel 4: pooled sums + counts per graph
// ---------------------------------------------------------------------------
__global__ void __launch_bounds__(256)
pool_kernel(const float* __restrict__ h2, const int* __restrict__ batch,
            float* __restrict__ sums, float* __restrict__ counts, int Nn) {
    long long stride = (long long)gridDim.x * blockDim.x;
    long long total = (long long)Nn * 32;
    for (long long i = (long long)blockIdx.x * blockDim.x + threadIdx.x; i < total; i += stride) {
        int n = (int)(i >> 5);
        int c = (int)(i & 31);
        int g = batch[n];
        float4 v = ((const float4*)(h2 + (size_t)n * HID_DIM))[c];
        float* drow = sums + (size_t)g * HID_DIM + (size_t)c * 4;
        atomicAdd(drow + 0, v.x);
        atomicAdd(drow + 1, v.y);
        atomicAdd(drow + 2, v.z);
        atomicAdd(drow + 3, v.w);
        if (c == 0) atomicAdd(counts + g, 1.0f);
    }
}

// ---------------------------------------------------------------------------
// Kernel 5: head  out = clip(relu(pooled@fc1w+fc1b)@fc2w+fc2b, -10, 10)
// ---------------------------------------------------------------------------
__global__ void __launch_bounds__(128)
head_kernel(const float* __restrict__ sums, const float* __restrict__ counts,
            const float* __restrict__ fc1w, const float* __restrict__ fc1b,
            const float* __restrict__ fc2w, const float* __restrict__ fc2b,
            float* __restrict__ out) {
    __shared__ float sp[HID_DIM];
    __shared__ float sh[HID_DIM];
    int g = blockIdx.x, t = threadIdx.x;
    float cnt = fmaxf(counts[g], 1.0f);
    sp[t] = sums[(size_t)g * HID_DIM + t] / cnt;
    __syncthreads();
    float acc = fc1b[t];
    for (int k = 0; k < HID_DIM; ++k) acc += sp[k] * fc1w[k * HID_DIM + t];
    sh[t] = fmaxf(acc, 0.0f);
    __syncthreads();
    if (t < NCLS) {
        float o = fc2b[t];
        for (int k = 0; k < HID_DIM; ++k) o += sh[k] * fc2w[k * NCLS + t];
        o = fminf(fmaxf(o, -10.0f), 10.0f);
        out[g * NCLS + t] = o;
    }
}

extern "C" void kernel_launch(void* const* d_in, const int* in_sizes, int n_in,
                              void* d_out, int out_size, void* d_ws, size_t ws_size,
                              hipStream_t stream) {
    const float* x      = (const float*)d_in[0];
    const float* topo   = (const float*)d_in[1];
    const int*   ei     = (const int*)d_in[2];
    const int*   batch  = (const int*)d_in[3];
    const float* w_lin  = (const float*)d_in[4];
    const float* b_lin  = (const float*)d_in[5];
    const float* w_gate = (const float*)d_in[6];
    const float* b_gate = (const float*)d_in[7];
    const float* w_tout = (const float*)d_in[8];
    const float* b_tout = (const float*)d_in[9];
    const float* c1w1 = (const float*)d_in[10];
    const float* c1b1 = (const float*)d_in[11];
    const float* c1w2 = (const float*)d_in[12];
    const float* c1b2 = (const float*)d_in[13];
    const float* c2w1 = (const float*)d_in[14];
    const float* c2b1 = (const float*)d_in[15];
    const float* c2w2 = (const float*)d_in[16];
    const float* c2b2 = (const float*)d_in[17];
    const float* fc1w = (const float*)d_in[18];
    const float* fc1b = (const float*)d_in[19];
    const float* fc2w = (const float*)d_in[20];
    const float* fc2b = (const float*)d_in[21];

    int Nn = in_sizes[0] / IN_DIM;
    int E  = in_sizes[2] / 2;
    int G  = out_size / NCLS;
    const int* src = ei;
    const int* dst = ei + E;

    // workspace: 3 x [N,128] f32 + pooled sums/counts + swizzled bf16 weights
    float* buf0 = (float*)d_ws;                     // xc, later h2
    float* buf1 = buf0 + (size_t)Nn * HID_DIM;      // agg (zeroed before each scatter)
    float* buf2 = buf1 + (size_t)Nn * HID_DIM;      // h1
    float* sums = buf2 + (size_t)Nn * HID_DIM;      // [G,128]
    float* counts = sums + (size_t)G * HID_DIM;     // [G]
    size_t wbytes = ((size_t)(3 * (size_t)Nn * HID_DIM + (size_t)G * HID_DIM + G) * 4 + 63) & ~(size_t)63;
    unsigned short* wswGated = (unsigned short*)((char*)d_ws + wbytes);   // 32 frags
    unsigned short* wswGin1  = wswGated + 32 * FRAG_ELEMS;                // 64 frags
    unsigned short* wswGin2  = wswGin1 + 64 * FRAG_ELEMS;                 // 64 frags

    int ntiles = (Nn + 15) / 16;
    int mm_blocks = (ntiles + 7) / 8;
    if (mm_blocks > 256) mm_blocks = 256;           // persistent-ish: amortize weight staging

    size_t gated_smem = (size_t)(32 * FRAG_ELEMS + 8 * 16 * HID_DIM) * 2
                      + (size_t)(2 * HID_DIM + IN_DIM) * 4;               // ~66KB
    size_t gin_smem = (size_t)(64 * FRAG_ELEMS + 8 * 16 * HID_DIM) * 2
                    + (size_t)(2 * HID_DIM) * 4;                          // ~97KB (<320KB/WGP)

    // pre-swizzle all weights into WMMA B-fragment layout (bf16)
    swizzle_w_kernel<<<8, 32, 0, stream>>>(w_lin, HID_DIM, wswGated);
    swizzle_w_kernel<<<8, 32, 0, stream>>>(w_gate, HID_DIM, wswGated + 8 * FRAG_ELEMS);
    swizzle_w_kernel<<<16, 32, 0, stream>>>(w_tout, IN_DIM, wswGated + 16 * FRAG_ELEMS);
    swizzle_w_kernel<<<32, 32, 0, stream>>>(c1w1, HID_DIM, wswGin1);
    swizzle_w_kernel<<<32, 32, 0, stream>>>(c1w2, HID_DIM, wswGin1 + 32 * FRAG_ELEMS);
    swizzle_w_kernel<<<32, 32, 0, stream>>>(c2w1, HID_DIM, wswGin2);
    swizzle_w_kernel<<<32, 32, 0, stream>>>(c2w2, HID_DIM, wswGin2 + 32 * FRAG_ELEMS);

    gated_topo_kernel<<<mm_blocks, 256, gated_smem, stream>>>(
        x, topo, wswGated, b_lin, b_gate, b_tout, buf0, Nn);

    hipMemsetAsync(buf1, 0, (size_t)Nn * HID_DIM * sizeof(float), stream);
    edge_scatter_kernel<<<8192, 256, 0, stream>>>(buf0, src, dst, buf1, E);
    gin_mlp_kernel<<<mm_blocks, 256, gin_smem, stream>>>(buf0, buf1, wswGin1, c1b1, c1b2, buf2, Nn);

    hipMemsetAsync(buf1, 0, (size_t)Nn * HID_DIM * sizeof(float), stream);
    edge_scatter_kernel<<<8192, 256, 0, stream>>>(buf2, src, dst, buf1, E);
    gin_mlp_kernel<<<mm_blocks, 256, gin_smem, stream>>>(buf2, buf1, wswGin2, c2b1, c2b2, buf0, Nn);

    hipMemsetAsync(sums, 0, ((size_t)G * HID_DIM + G) * sizeof(float), stream);
    pool_kernel<<<4096, 256, 0, stream>>>(buf0, batch, sums, counts, Nn);
    head_kernel<<<G, 128, 0, stream>>>(sums, counts, fc1w, fc1b, fc2w, fc2b, (float*)d_out);
}
